// MultiSourceAttentionBlock_83614423318762
// MI455X (gfx1250) — matrix operand order repaired and checked by reading, not compiled, not run on hardware
//
#include <hip/hip_runtime.h>
#include <hip/hip_bf16.h>

typedef __attribute__((ext_vector_type(16))) _Float16 v16h;
typedef __attribute__((ext_vector_type(8)))  _Float16 v8h;
typedef __attribute__((ext_vector_type(8)))  float    v8f;

// ---------------------------------------------------------------------------
// WMMA helpers (gfx1250, wave32). D = A(16xK) * B(Kx16) + C, K=32, f16 in, f32 acc.
// A-operand: lane(0-15)=row M, holds K 0..15;
//            lane(16-31)=row M, holds K 16..31.   -> one contiguous 32B load
// B-operand: lane(&15)=col N, lanes 0-15 hold K 0..15, lanes 16-31 K 16..31
//            -> one contiguous 32B load per lane.
// ---------------------------------------------------------------------------
__device__ __forceinline__ v8f wmma16(v16h a, v16h b, v8f c) {
  return __builtin_amdgcn_wmma_f32_16x16x32_f16(false, a, false, b, (short)0, c,
                                                false, false);
}

__device__ __forceinline__ v16h load_fragA(const _Float16* base, int ld, int lane) {
  int r = lane & 15, hh = lane >> 4;
  const _Float16* p = base + (size_t)r * ld + hh * 16;
  return *(const v16h*)p;
}

__device__ __forceinline__ v16h load_fragB(const _Float16* base, int ld, int lane) {
  int c = lane & 15, hh = lane >> 4;
  const _Float16* p = base + (size_t)c * ld + hh * 16;
  return *(const v16h*)p;
}

// Per-lane 16B async DMA global -> LDS (ASYNCcnt domain, overlaps with WMMA).
__device__ __forceinline__ void async_g2l_b128(const _Float16* g, _Float16* l) {
  unsigned loff = (unsigned)(size_t)(void*)l;  // low 32 bits = LDS offset (AS3)
  asm volatile("global_load_async_to_lds_b128 %0, %1, off"
               :: "v"(loff), "v"(g) : "memory");
}

// ---------------------------------------------------------------------------
// Generic GEMM: C[M,N] = A[M,K] @ W[N,K]^T + bias, optional exact-GELU,
// f32 or f16 output. Block = 256 threads (8 waves), macro tile 128x64,
// each wave owns a 32x32 region (4 WMMA accumulators).
// W (B-operand) tile (64 rows x 32 K-halves = 4KB) is staged to LDS with
// double-buffered GLOBAL_LOAD_ASYNC_TO_LDS_B128: 4x reuse across wave-rows,
// DMA overlapped with WMMA on the other buffer.
// ---------------------------------------------------------------------------
__global__ __launch_bounds__(256)
void gemm_wmma_kernel(const _Float16* __restrict__ A,
                      const _Float16* __restrict__ W,
                      const float* __restrict__ bias,
                      float* __restrict__ c32, _Float16* __restrict__ c16,
                      int M, int N, int K, int act_gelu) {
  __shared__ __align__(32) _Float16 Wl[2][64 * 32];

  int tid  = threadIdx.x;
  int lane = tid & 31;
  int w    = tid >> 5;
  int wr   = w & 3;           // 4 wave-rows of 32
  int wc   = w >> 2;          // 2 wave-cols of 32
  int m0   = blockIdx.y * 128 + wr * 32;
  int nblk = blockIdx.x * 64;
  int n0   = nblk + wc * 32;

  const _Float16* A0 = A + (size_t)m0 * K;
  const _Float16* A1 = A + (size_t)(m0 + 16) * K;

  // cooperative staging coords: 256 threads cover 64 rows x 4 x 16B
  int srow = tid >> 2, sseg = tid & 3;
  const _Float16* gW = W + (size_t)(nblk + srow) * K + sseg * 8;
  _Float16* lW = &Wl[0][srow * 32 + sseg * 8];

  v8f acc[2][2] = {};
  async_g2l_b128(gW, lW);                       // stage buffer 0, k0 = 0
  for (int k0 = 0; k0 < K; k0 += 32) {
    int cur = (k0 >> 5) & 1;
    if (k0 + 32 < K) {
      async_g2l_b128(gW + (k0 + 32), &Wl[cur ^ 1][srow * 32 + sseg * 8]);
      asm volatile("s_wait_asynccnt 0x1" ::: "memory");  // cur done, next in flight
    } else {
      asm volatile("s_wait_asynccnt 0x0" ::: "memory");
    }
    __syncthreads();

    __builtin_prefetch(A0 + k0 + 128, 0, 1);
    v16h a0 = load_fragA(A0 + k0, K, lane);
    v16h a1 = load_fragA(A1 + k0, K, lane);
    const _Float16* Wb = &Wl[cur][(wc * 32) * 32];
    v16h b0 = load_fragB(Wb, 32, lane);
    v16h b1 = load_fragB(Wb + 16 * 32, 32, lane);

    acc[0][0] = wmma16(a0, b0, acc[0][0]);
    acc[0][1] = wmma16(a0, b1, acc[0][1]);
    acc[1][0] = wmma16(a1, b0, acc[1][0]);
    acc[1][1] = wmma16(a1, b1, acc[1][1]);
    __syncthreads();                            // all waves done with cur buffer
  }

  int rl = lane & 15, hh = lane >> 4;
#pragma unroll
  for (int ti = 0; ti < 2; ++ti)
#pragma unroll
    for (int tj = 0; tj < 2; ++tj)
#pragma unroll
      for (int rr = 0; rr < 8; ++rr) {
        int m = m0 + ti * 16 + hh * 8 + rr;
        int n = n0 + tj * 16 + rl;
        float v = acc[ti][tj][rr] + bias[n];
        if (act_gelu) v = 0.5f * v * (1.0f + erff(v * 0.70710678118f));
        if (c16) c16[(size_t)m * N + n] = (_Float16)v;
        else     c32[(size_t)m * N + n] = v;
      }
}

// ---------------------------------------------------------------------------
// Flash attention over 4 key/value sources with joint (online) softmax.
// One wave per (b, h, 16-query tile). S = Q(16x64) Kt(64x32) via 4 WMMAs,
// P(16x32) staged via LDS, O += P @ V via 4 WMMAs (V pre-transposed (B,H,64,L)).
// ---------------------------------------------------------------------------
__global__ __launch_bounds__(32)
void flash_attn_kernel(const _Float16* __restrict__ Q0, const _Float16* __restrict__ Q1,
                       const _Float16* __restrict__ Q2, const _Float16* __restrict__ Q3,
                       const _Float16* __restrict__ K0, const _Float16* __restrict__ K1,
                       const _Float16* __restrict__ K2, const _Float16* __restrict__ K3,
                       const _Float16* __restrict__ V0, const _Float16* __restrict__ V1,
                       const _Float16* __restrict__ V2, const _Float16* __restrict__ V3,
                       const float* __restrict__ temps, const float* __restrict__ sbias,
                       const float* __restrict__ gating, _Float16* __restrict__ O16) {
  __shared__ __align__(32) _Float16 Plds[16 * 32];
  const _Float16* Qs[4] = {Q0, Q1, Q2, Q3};
  const _Float16* Ks[4] = {K0, K1, K2, K3};
  const _Float16* Vs[4] = {V0, V1, V2, V3};
  const int Ls[4] = {512, 512, 512, 256};

  int lane = threadIdx.x;
  int rl = lane & 15, hh = lane >> 4;
  int qt0 = blockIdx.x * 16;
  int bh  = blockIdx.y;
  int b = bh >> 4, h = bh & 15;

  float ratio = tanhf(gating[0]);
  float mrow[8], lrow[8];
  v8f acc[4] = {};
#pragma unroll
  for (int r = 0; r < 8; ++r) { mrow[r] = -1e30f; lrow[r] = 0.0f; }

#pragma unroll
  for (int src = 0; src < 4; ++src) {
    const int L = Ls[src];
    const _Float16* Qp = Qs[src] + ((size_t)bh * 512 + qt0) * 64;
    const _Float16* Kp = Ks[src] + (size_t)bh * L * 64;
    const _Float16* Vp = Vs[src] + (size_t)bh * 64 * L;
    v16h q0 = load_fragA(Qp, 64, lane);        // d = 0..31
    v16h q1 = load_fragA(Qp + 32, 64, lane);   // d = 32..63
    float tempv = 1.0f + log1pf(expf(temps[src]));
    float sc = 0.125f * tempv;                 // HD^-0.5 * temp
    float sb = sbias[src];
    float gmul = (src == 3) ? ratio : 1.0f;

    for (int kb = 0; kb < L; kb += 32) {
      v16h k00 = load_fragB(Kp + (size_t)kb * 64, 64, lane);
      v16h k01 = load_fragB(Kp + (size_t)kb * 64 + 32, 64, lane);
      v16h k10 = load_fragB(Kp + (size_t)(kb + 16) * 64, 64, lane);
      v16h k11 = load_fragB(Kp + (size_t)(kb + 16) * 64 + 32, 64, lane);
      v8f s0 = {}, s1 = {};
      s0 = wmma16(q0, k00, s0);  s0 = wmma16(q1, k01, s0);
      s1 = wmma16(q0, k10, s1);  s1 = wmma16(q1, k11, s1);

      float p0a[8], p1a[8], alpha[8];
#pragma unroll
      for (int r = 0; r < 8; ++r) {
        float a0 = (s0[r] * sc + sb) * gmul;
        float a1 = (s1[r] * sc + sb) * gmul;
        float cm = fmaxf(a0, a1);
        cm = fmaxf(cm, __shfl_xor(cm, 1));
        cm = fmaxf(cm, __shfl_xor(cm, 2));
        cm = fmaxf(cm, __shfl_xor(cm, 4));
        cm = fmaxf(cm, __shfl_xor(cm, 8));
        float mn = fmaxf(mrow[r], cm);
        alpha[r] = __expf(mrow[r] - mn);
        mrow[r] = mn;
        p0a[r] = __expf(a0 - mn);
        p1a[r] = __expf(a1 - mn);
        float rs = p0a[r] + p1a[r];
        rs += __shfl_xor(rs, 1);
        rs += __shfl_xor(rs, 2);
        rs += __shfl_xor(rs, 4);
        rs += __shfl_xor(rs, 8);
        lrow[r] = lrow[r] * alpha[r] + rs;
      }
#pragma unroll
      for (int t = 0; t < 4; ++t)
#pragma unroll
        for (int r = 0; r < 8; ++r) acc[t][r] *= alpha[r];

#pragma unroll
      for (int r = 0; r < 8; ++r) {
        Plds[(hh * 8 + r) * 32 + rl]      = (_Float16)p0a[r];
        Plds[(hh * 8 + r) * 32 + rl + 16] = (_Float16)p1a[r];
      }
      __syncthreads();
      v16h pf = load_fragA(Plds, 32, lane);
#pragma unroll
      for (int t = 0; t < 4; ++t) {
        v16h vf = load_fragB(Vp + (size_t)(t * 16) * L + kb, L, lane);
        acc[t] = wmma16(pf, vf, acc[t]);
      }
      __syncthreads();
    }
  }

  float inv[8];
#pragma unroll
  for (int r = 0; r < 8; ++r) inv[r] = 1.0f / lrow[r];
#pragma unroll
  for (int t = 0; t < 4; ++t)
#pragma unroll
    for (int r = 0; r < 8; ++r) {
      size_t oi = ((size_t)(b * 512 + qt0 + hh * 8 + r)) * 1024 + h * 64 + t * 16 + rl;
      O16[oi] = (_Float16)(acc[t][r] * inv[r]);
    }
}

// ---------------------------------------------------------------------------
// Elementwise / reduction kernels
// ---------------------------------------------------------------------------
__global__ void f32_to_f16_kernel(const float* __restrict__ s,
                                  _Float16* __restrict__ d, int n) {
  int i = blockIdx.x * blockDim.x + threadIdx.x;
  if (i < n) d[i] = (_Float16)s[i];
}

__global__ void silu_kernel(const float* __restrict__ c, float* __restrict__ o, int n) {
  int i = blockIdx.x * blockDim.x + threadIdx.x;
  if (i < n) { float v = c[i]; o[i] = v / (1.0f + expf(-v)); }
}

// mod[b,j] = silu(cond[b,:]) . adaln_w[j,:] + adaln_b[j]   (B x 6D)
__global__ void adaln_dot_kernel(const float* __restrict__ sc,
                                 const float* __restrict__ wA,
                                 const float* __restrict__ bA,
                                 float* __restrict__ mod) {
  int i = blockIdx.x * blockDim.x + threadIdx.x;
  if (i >= 4 * 6144) return;
  int b = i / 6144, j = i % 6144;
  const float* s = sc + b * 1024;
  const float* wr = wA + (size_t)j * 1024;
  float acc = bA[j];
  for (int d = 0; d < 1024; ++d) acc += s[d] * wr[d];
  mod[i] = acc;
}

// LayerNorm + AdaLN modulate -> f16. One block per token.
__global__ __launch_bounds__(256)
void ln_mod_kernel(const float* __restrict__ X, const float* __restrict__ w,
                   const float* __restrict__ bb, const float* __restrict__ mod,
                   int shift_idx, int scale_idx, _Float16* __restrict__ out16, int Tv) {
  __shared__ float s1[256], s2[256];
  int tok = blockIdx.x;
  int b = tok / Tv;
  const float* row = X + (size_t)tok * 1024;
  float a = 0.0f, sq = 0.0f;
  for (int j = threadIdx.x; j < 1024; j += 256) { float v = row[j]; a += v; sq += v * v; }
  s1[threadIdx.x] = a; s2[threadIdx.x] = sq;
  __syncthreads();
  for (int off = 128; off > 0; off >>= 1) {
    if ((int)threadIdx.x < off) { s1[threadIdx.x] += s1[threadIdx.x + off];
                                  s2[threadIdx.x] += s2[threadIdx.x + off]; }
    __syncthreads();
  }
  float mean = s1[0] * (1.0f / 1024.0f);
  float var  = s2[0] * (1.0f / 1024.0f) - mean * mean;
  float inv  = rsqrtf(var + 1e-5f);
  const float* mrow = mod + (size_t)b * 6144;
  for (int j = threadIdx.x; j < 1024; j += 256) {
    float y = (row[j] - mean) * inv * w[j] + bb[j];
    y = y * (1.0f + mrow[scale_idx * 1024 + j]) + mrow[shift_idx * 1024 + j];
    out16[(size_t)tok * 1024 + j] = (_Float16)y;
  }
}

// RMS-norm per head (+optional RoPE), pack (B,L,D)f32 -> (B,H,L,64)f16.
__global__ __launch_bounds__(256)
void rms_rope_pack_kernel(const float* __restrict__ raw, const float* __restrict__ rw,
                          _Float16* __restrict__ packed, int Lv, int do_rope) {
  __shared__ float ss[256];
  int tid = threadIdx.x;
  int g = blockIdx.x * 4 + (tid >> 6);
  int d = tid & 63;
  int hh = g % 16;
  int l  = (g / 16) % Lv;
  int b  = g / (16 * Lv);
  size_t rbase = ((size_t)(b * Lv + l)) * 1024 + hh * 64;
  float xv = raw[rbase + d];
  ss[tid] = xv * xv;
  __syncthreads();
  for (int off = 32; off > 0; off >>= 1) {
    if ((tid & 63) < off) ss[tid] += ss[tid + off];
    __syncthreads();
  }
  float inv = rsqrtf(ss[tid & ~63] * (1.0f / 64.0f) + 1e-6f);
  float y = xv * inv * rw[d];
  if (do_rope) {
    int dp = (d + 32) & 63;
    float yp = raw[rbase + dp] * inv * rw[dp];
    float rot = (d < 32) ? -yp : yp;
    float fr = __powf(10000.0f, -(float)(d & 31) * (1.0f / 32.0f));
    float e = (float)l * fr;
    y = y * __cosf(e) + rot * __sinf(e);
  }
  packed[(((size_t)(b * 16 + hh)) * Lv + l) * 64 + d] = (_Float16)y;
}

// Pack V: (B,L,D)f32 -> transposed (B,H,64,L)f16
__global__ void vpack_kernel(const float* __restrict__ raw, _Float16* __restrict__ vt,
                             int Lv, int total) {
  int idx = blockIdx.x * blockDim.x + threadIdx.x;
  if (idx >= total) return;
  int d  = idx & 63;
  int hh = (idx >> 6) & 15;
  int tok = idx >> 10;
  int l = tok % Lv, b = tok / Lv;
  vt[(((size_t)(b * 16 + hh)) * 64 + d) * Lv + l] = (_Float16)raw[idx];
}

// out = x + mod[b, gate_idx*D + d] * y
__global__ void res_gate_kernel(const float* __restrict__ xin, const float* __restrict__ y,
                                const float* __restrict__ mod, int gate_idx,
                                float* __restrict__ out, int TD, int n) {
  int i = blockIdx.x * blockDim.x + threadIdx.x;
  if (i >= n) return;
  int b = i / TD, d = i % 1024;
  out[i] = xin[i] + mod[(size_t)b * 6144 + gate_idx * 1024 + d] * y[i];
}

// ---------------------------------------------------------------------------
// Host orchestration
// ---------------------------------------------------------------------------
extern "C" void kernel_launch(void* const* d_in, const int* in_sizes, int n_in,
                              void* d_out, int out_size, void* d_ws, size_t ws_size,
                              hipStream_t stream) {
  (void)in_sizes; (void)n_in; (void)out_size; (void)ws_size;
  const float* x        = (const float*)d_in[0];
  const float* bev      = (const float*)d_in[1];
  const float* vl       = (const float*)d_in[2];
  const float* rea      = (const float*)d_in[3];
  const float* cond     = (const float*)d_in[4];
  const float* ln_pre_w = (const float*)d_in[5];
  const float* ln_pre_b = (const float*)d_in[6];
  const float* adaln_w  = (const float*)d_in[7];
  const float* adaln_b  = (const float*)d_in[8];
  const float* q_proj_w = (const float*)d_in[9];
  const float* q_proj_b = (const float*)d_in[10];
  const float* qa_w     = (const float*)d_in[11];
  const float* qa_b     = (const float*)d_in[12];
  const float* qa_out_w = (const float*)d_in[13];
  const float* qa_out_b = (const float*)d_in[14];
  const float* k_w      = (const float*)d_in[15];
  const float* k_b      = (const float*)d_in[16];
  const float* v_w      = (const float*)d_in[17];
  const float* v_b      = (const float*)d_in[18];
  const float* o_w      = (const float*)d_in[19];
  const float* o_b      = (const float*)d_in[20];
  const float* qn_w     = (const float*)d_in[21];
  const float* kn_w     = (const float*)d_in[22];
  const float* gating   = (const float*)d_in[23];
  const float* temps    = (const float*)d_in[24];
  const float* sbias    = (const float*)d_in[25];
  const float* ffn_ln_w = (const float*)d_in[26];
  const float* ffn_ln_b = (const float*)d_in[27];
  const float* ffn_w1   = (const float*)d_in[28];
  const float* ffn_b1   = (const float*)d_in[29];
  const float* ffn_w2   = (const float*)d_in[30];
  const float* ffn_b2   = (const float*)d_in[31];

  const int B = 4, T = 512, D = 1024, H = 16, D4 = 256, DFF = 4096;
  const int BT = B * T;                 // 2048 tokens
  const int Lsrc[4] = {512, 512, 512, 256};

  char* ws = (char*)d_ws;
  size_t off = 0;
  auto a16 = [&](size_t n) -> _Float16* {
    _Float16* p = (_Float16*)(ws + off);
    off += ((n * sizeof(_Float16) + 255) & ~(size_t)255);
    return p;
  };
  auto a32 = [&](size_t n) -> float* {
    float* p = (float*)(ws + off);
    off += ((n * sizeof(float) + 255) & ~(size_t)255);
    return p;
  };

  // f16 weights
  _Float16* w16_q   = a16((size_t)D * D);
  _Float16* w16_k   = a16((size_t)4 * D * D);
  _Float16* w16_v   = a16((size_t)4 * D * D);
  _Float16* w16_qa  = a16((size_t)3 * D4 * D);
  _Float16* w16_qao = a16((size_t)D * D4);
  _Float16* w16_o   = a16((size_t)D * D);
  _Float16* w16_f1  = a16((size_t)DFF * D);
  _Float16* w16_f2  = a16((size_t)D * DFF);
  // f16 activations
  _Float16* xn16  = a16((size_t)BT * D);
  _Float16* bev16 = a16((size_t)BT * D);
  _Float16* vl16  = a16((size_t)BT * D);
  _Float16* re16  = a16((size_t)B * 256 * D);
  _Float16* hid16 = a16((size_t)3 * BT * D4);
  _Float16* qpk[4]; for (int i = 0; i < 4; ++i) qpk[i] = a16((size_t)BT * D);
  _Float16* kpk[4]; for (int i = 0; i < 4; ++i) kpk[i] = a16((size_t)B * Lsrc[i] * D);
  _Float16* vtp[4]; for (int i = 0; i < 4; ++i) vtp[i] = a16((size_t)B * Lsrc[i] * D);
  _Float16* o16 = a16((size_t)BT * D);
  _Float16* h16 = a16((size_t)BT * D);
  _Float16* g16 = a16((size_t)BT * DFF);
  // f32 scratch
  float* siluc = a32((size_t)B * D);
  float* modv  = a32((size_t)B * 6 * D);
  float* raw   = a32((size_t)BT * D);     // reused for every raw GEMM output
  float* aproj = a32((size_t)BT * D);
  float* x2    = a32((size_t)BT * D);

  auto cvt = [&](const float* s, _Float16* dd, size_t n) {
    int nn = (int)n;
    f32_to_f16_kernel<<<(nn + 255) / 256, 256, 0, stream>>>(s, dd, nn);
  };
  auto gemm = [&](const _Float16* A, const _Float16* W, const float* bias,
                  float* c32, _Float16* c16, int M, int N, int K, int act) {
    dim3 grid(N / 64, M / 128);
    gemm_wmma_kernel<<<grid, 256, 0, stream>>>(A, W, bias, c32, c16, M, N, K, act);
  };

  // 1) weight + token conversions to f16
  cvt(q_proj_w, w16_q,  (size_t)D * D);
  cvt(k_w,      w16_k,  (size_t)4 * D * D);
  cvt(v_w,      w16_v,  (size_t)4 * D * D);
  cvt(qa_w,     w16_qa, (size_t)3 * D4 * D);
  cvt(qa_out_w, w16_qao,(size_t)D * D4);
  cvt(o_w,      w16_o,  (size_t)D * D);
  cvt(ffn_w1,   w16_f1, (size_t)DFF * D);
  cvt(ffn_w2,   w16_f2, (size_t)D * DFF);
  cvt(bev, bev16, (size_t)BT * D);
  cvt(vl,  vl16,  (size_t)BT * D);
  cvt(rea, re16,  (size_t)B * 256 * D);

  // 2) AdaLN modulation
  silu_kernel<<<(B * D + 255) / 256, 256, 0, stream>>>(cond, siluc, B * D);
  adaln_dot_kernel<<<(B * 6 * D + 255) / 256, 256, 0, stream>>>(siluc, adaln_w, adaln_b, modv);

  // 3) pre-LN + modulate  (shift_pre=chunk0, scale_pre=chunk1)
  ln_mod_kernel<<<BT, 256, 0, stream>>>(x, ln_pre_w, ln_pre_b, modv, 0, 1, xn16, T);

  // 4) self Q: GEMM + RMS + RoPE pack
  gemm(xn16, w16_q, q_proj_b, raw, nullptr, BT, D, D, 0);
  rms_rope_pack_kernel<<<BT * H / 4, 256, 0, stream>>>(raw, qn_w, qpk[0], T, 1);

  // 5) adapter queries: D -> D/4 -> D, RMS (no RoPE)
  for (int i = 0; i < 3; ++i) {
    gemm(xn16, w16_qa + (size_t)i * D4 * D, qa_b + i * D4,
         nullptr, hid16 + (size_t)i * BT * D4, BT, D4, D, 0);
    gemm(hid16 + (size_t)i * BT * D4, w16_qao, qa_out_b, raw, nullptr, BT, D, D4, 0);
    rms_rope_pack_kernel<<<BT * H / 4, 256, 0, stream>>>(raw, qn_w, qpk[i + 1], T, 0);
  }

  // 6) K/V per source
  const _Float16* srcs16[4] = {xn16, bev16, vl16, re16};
  for (int i = 0; i < 4; ++i) {
    int Mi = B * Lsrc[i];
    gemm(srcs16[i], w16_k + (size_t)i * D * D, k_b + i * D, raw, nullptr, Mi, D, D, 0);
    rms_rope_pack_kernel<<<Mi * H / 4, 256, 0, stream>>>(raw, kn_w, kpk[i], Lsrc[i], 1);
    gemm(srcs16[i], w16_v + (size_t)i * D * D, v_b + i * D, raw, nullptr, Mi, D, D, 0);
    int tot = Mi * D;
    vpack_kernel<<<(tot + 255) / 256, 256, 0, stream>>>(raw, vtp[i], Lsrc[i], tot);
  }

  // 7) joint-softmax flash attention -> O (token-major f16)
  flash_attn_kernel<<<dim3(T / 16, B * H), 32, 0, stream>>>(
      qpk[0], qpk[1], qpk[2], qpk[3],
      kpk[0], kpk[1], kpk[2], kpk[3],
      vtp[0], vtp[1], vtp[2], vtp[3],
      temps, sbias, gating, o16);

  // 8) output projection + gated residual (gate_attn = chunk2)
  gemm(o16, w16_o, o_b, aproj, nullptr, BT, D, D, 0);
  res_gate_kernel<<<(BT * D + 255) / 256, 256, 0, stream>>>(x, aproj, modv, 2, x2, T * D, BT * D);

  // 9) FFN: LN(mod chunks 3,4) -> W1+GELU -> W2 -> gated residual (chunk5)
  ln_mod_kernel<<<BT, 256, 0, stream>>>(x2, ffn_ln_w, ffn_ln_b, modv, 3, 4, h16, T);
  gemm(h16, w16_f1, ffn_b1, nullptr, g16, BT, DFF, D, 1);
  gemm(g16, w16_f2, ffn_b2, raw, nullptr, BT, D, DFF, 0);
  res_gate_kernel<<<(BT * D + 255) / 256, 256, 0, stream>>>(x2, raw, modv, 5,
                                                            (float*)d_out, T * D, BT * D);
}
